// DeformableConv2d_28449863369523
// MI455X (gfx1250) — matrix operand order repaired and checked by reading, not compiled, not run on hardware
//
#include <hip/hip_runtime.h>
#include <hip/hip_fp16.h>

typedef __attribute__((ext_vector_type(16))) _Float16 v16h;
typedef __attribute__((ext_vector_type(8)))  float    v8f;

#define C_IN  64
#define H_DIM 128
#define W_DIM 128
#define HW    (H_DIM * W_DIM)
#define KKN   9
#define KDIM  (C_IN * KKN)   // 576
#define NKB   (KDIM / 32)    // 18 K-blocks of 32
#define COUT  64
#define OFFC  18

union AFrag { v16h v; _Float16 h[16]; unsigned u[8]; uint4 q[2]; };
union DFrag { v8f  v; float f[8]; };

__device__ inline unsigned pack_h2(float a, float b) {
  union { _Float16 h[2]; unsigned u; } p;
  p.h[0] = (_Float16)a; p.h[1] = (_Float16)b;
  return p.u;
}

// ---------------------------------------------------------------------------
// Pack weights into per-lane WMMA A-fragment order (16-bit A 16x32 layout),
// split into f16 hi + f16 residual lo terms.
// K ordering: k = kk*64 + c.  e = (((term*MT + m)*NKB + kb)*32 + lane)*8 + v
// ---------------------------------------------------------------------------
__global__ void prep_pack_weights(const float* __restrict__ dw,
                                  const float* __restrict__ ow,
                                  unsigned* __restrict__ packD,
                                  unsigned* __restrict__ packO) {
  const int ND = 2 * 4 * NKB * 32 * 8;
  const int NO = 2 * 2 * NKB * 32 * 8;
  int tid = blockIdx.x * blockDim.x + threadIdx.x;
  if (tid >= ND + NO) return;
  bool isO = tid >= ND;
  int e = isO ? (tid - ND) : tid;
  int v    = e & 7;
  int lane = (e >> 3) & 31;
  int kb   = (e >> 8) % NKB;
  int rest = (e >> 8) / NKB;
  int mt   = isO ? (rest & 1) : (rest & 3);
  int term = isO ? (rest >> 1) : (rest >> 2);
  int row  = mt * 16 + (lane & 15);

  float vals[2];
  #pragma unroll
  for (int j = 0; j < 2; ++j) {
    int kl = (lane >> 4) * 16 + 2*v + j;
    int k  = kb * 32 + kl;
    int kk = k >> 6;
    int c  = k & 63;
    float w;
    if (isO) w = (row < OFFC) ? ow[(row * C_IN + c) * KKN + kk] : 0.0f;
    else     w = dw[(row * C_IN + c) * KKN + kk];
    float hi = (float)(_Float16)w;
    vals[j] = (term == 0) ? w : (w - hi);
  }
  unsigned out = pack_h2(vals[0], vals[1]);
  if (isO) packO[e] = out; else packD[e] = out;
}

// ---------------------------------------------------------------------------
// NCHW -> NHWC transpose of x via LDS tile (stride 65 padding: conflict-free),
// both global sides coalesced. One-time ~67 MB round trip.
// ---------------------------------------------------------------------------
__global__ __launch_bounds__(256) void transpose_nchw_nhwc(
    const float* __restrict__ x, float* __restrict__ xt) {
  __shared__ float lds[64 * 65];
  int bh = blockIdx.x;                 // b*H + y
  int b  = bh / H_DIM;
  int y  = bh % H_DIM;
  int w0 = blockIdx.y * 64;
  const float* xrow = x + ((size_t)b * C_IN * H_DIM + y) * W_DIM + w0;
  #pragma unroll
  for (int it = 0; it < 16; ++it) {
    int idx = it * 256 + threadIdx.x;
    int c = idx >> 6;                  // consecutive threads -> consecutive w
    int w = idx & 63;
    lds[w * 65 + c] = xrow[(size_t)c * HW + w];
  }
  __syncthreads();
  float* orow = xt + ((size_t)bh * W_DIM + w0) * C_IN;
  #pragma unroll
  for (int it = 0; it < 16; ++it) {
    int idx = it * 256 + threadIdx.x;
    int w = idx >> 6;                  // consecutive threads -> consecutive c
    int c = idx & 63;
    orow[(size_t)w * C_IN + c] = lds[w * 65 + c];
  }
}

// ---------------------------------------------------------------------------
// Offset conv (3x3, 64->18) as implicit GEMM, NHWC input: 4 x b128 per block.
// ---------------------------------------------------------------------------
__global__ __launch_bounds__(128) void offset_gemm(
    const float* __restrict__ xt, const unsigned* __restrict__ packO,
    const float* __restrict__ ob, float* __restrict__ off, int Bn) {
  int lane  = threadIdx.x & 31;
  int tile  = blockIdx.x * 4 + (threadIdx.x >> 5);
  const int tpr = W_DIM / 16;
  int w0 = (tile % tpr) * 16;
  int h  = (tile / tpr) % H_DIM;
  int b  = tile / (tpr * H_DIM);
  if (b >= Bn) return;                 // uniform per wave
  int px    = lane & 15;
  int khalf = lane >> 4;
  const float* xtb = xt + (size_t)b * HW * C_IN;

  DFrag acc[2];
  #pragma unroll
  for (int m = 0; m < 2; ++m)
    #pragma unroll
    for (int r = 0; r < 8; ++r) acc[m].f[r] = 0.0f;

  for (int kk = 0; kk < KKN; ++kk) {
    int ky = kk / 3, kx = kk % 3;
    int iy = h + ky - 1;
    int ix = w0 + px + kx - 1;
    float msk = (iy >= 0 && iy < H_DIM && ix >= 0 && ix < W_DIM) ? 1.0f : 0.0f;
    int iyc = min(max(iy, 0), H_DIM - 1);
    int ixc = min(max(ix, 0), W_DIM - 1);
    const float* base = xtb + (size_t)(iyc * W_DIM + ixc) * C_IN;

    for (int cb = 0; cb < 2; ++cb) {
      int kb = kk * 2 + cb;
      int cbase = cb * 32 + khalf * 16;
      const float4* p = (const float4*)(base + cbase);
      float val[16];
      #pragma unroll
      for (int qd = 0; qd < 4; ++qd) {
        float4 q = p[qd];
        val[qd*4+0] = msk * q.x; val[qd*4+1] = msk * q.y;
        val[qd*4+2] = msk * q.z; val[qd*4+3] = msk * q.w;
      }
      AFrag Bh, Bl;
      #pragma unroll
      for (int v = 0; v < 8; ++v) {
        float a0 = val[2*v], a1 = val[2*v+1];
        Bh.u[v] = pack_h2(a0, a1);
        Bl.u[v] = pack_h2(a0 - (float)(_Float16)a0, a1 - (float)(_Float16)a1);
      }
      #pragma unroll
      for (int m = 0; m < 2; ++m) {
        AFrag Ah, Al;
        const uint4* pH = (const uint4*)(packO + ((((size_t)(0*2 + m) * NKB + kb) * 32 + lane) * 8));
        const uint4* pL = (const uint4*)(packO + ((((size_t)(1*2 + m) * NKB + kb) * 32 + lane) * 8));
        Ah.q[0] = pH[0]; Ah.q[1] = pH[1];
        Al.q[0] = pL[0]; Al.q[1] = pL[1];
        acc[m].v = __builtin_amdgcn_wmma_f32_16x16x32_f16(false, Ah.v, false, Bh.v, (short)0, acc[m].v, false, false);
        acc[m].v = __builtin_amdgcn_wmma_f32_16x16x32_f16(false, Ah.v, false, Bl.v, (short)0, acc[m].v, false, false);
        acc[m].v = __builtin_amdgcn_wmma_f32_16x16x32_f16(false, Al.v, false, Bh.v, (short)0, acc[m].v, false, false);
      }
    }
  }
  #pragma unroll
  for (int m = 0; m < 2; ++m)
    #pragma unroll
    for (int r = 0; r < 8; ++r) {
      int o = m * 16 + r + khalf * 8;
      if (o < OFFC)
        off[(((size_t)b * OFFC + o) * H_DIM + h) * W_DIM + w0 + px] = acc[m].f[r] + ob[o];
    }
}

// ---------------------------------------------------------------------------
// Deformable conv as implicit GEMM, NHWC input: bilinear B-fill is
// 4 corners x 4 b128 loads per K-block (vs 64 scalar gathers in NCHW).
// ---------------------------------------------------------------------------
__global__ __launch_bounds__(128) void deform_gemm(
    const float* __restrict__ xt, const float* __restrict__ off,
    const unsigned* __restrict__ packD, const float* __restrict__ db,
    float* __restrict__ out, int Bn) {
  int lane  = threadIdx.x & 31;
  int tile  = blockIdx.x * 4 + (threadIdx.x >> 5);
  const int tpr = W_DIM / 16;
  int w0 = (tile % tpr) * 16;
  int h  = (tile / tpr) % H_DIM;
  int b  = tile / (tpr * H_DIM);
  if (b >= Bn) return;                 // uniform per wave
  int px    = lane & 15;
  int khalf = lane >> 4;
  const float* xtb  = xt  + (size_t)b * HW * C_IN;
  const float* offb = off + (size_t)b * OFFC * HW + h * W_DIM + (w0 + px);

  DFrag acc[4];
  #pragma unroll
  for (int m = 0; m < 4; ++m)
    #pragma unroll
    for (int r = 0; r < 8; ++r) acc[m].f[r] = 0.0f;

  for (int kk = 0; kk < KKN; ++kk) {
    int ky = kk / 3, kx = kk % 3;
    float oy = offb[(2 * kk + 0) * HW];
    float ox = offb[(2 * kk + 1) * HW];
    float ys = (float)(h + ky - 1) + oy;
    float xs = (float)(w0 + px + kx - 1) + ox;
    float y0f = floorf(ys), x0f = floorf(xs);
    float dy = ys - y0f, dx = xs - x0f;
    int y0 = (int)y0f, x0 = (int)x0f;
    int y1 = y0 + 1,  x1 = x0 + 1;
    float vy0 = (y0 >= 0 && y0 < H_DIM) ? 1.0f : 0.0f;
    float vy1 = (y1 >= 0 && y1 < H_DIM) ? 1.0f : 0.0f;
    float vx0 = (x0 >= 0 && x0 < W_DIM) ? 1.0f : 0.0f;
    float vx1 = (x1 >= 0 && x1 < W_DIM) ? 1.0f : 0.0f;
    int y0c = min(max(y0, 0), H_DIM - 1), y1c = min(max(y1, 0), H_DIM - 1);
    int x0c = min(max(x0, 0), W_DIM - 1), x1c = min(max(x1, 0), W_DIM - 1);
    float cw0 = (1.0f - dy) * (1.0f - dx) * vy0 * vx0;
    float cw1 = (1.0f - dy) * dx          * vy0 * vx1;
    float cw2 = dy * (1.0f - dx)          * vy1 * vx0;
    float cw3 = dy * dx                   * vy1 * vx1;
    const float* p0 = xtb + (size_t)(y0c * W_DIM + x0c) * C_IN;
    const float* p1 = xtb + (size_t)(y0c * W_DIM + x1c) * C_IN;
    const float* p2 = xtb + (size_t)(y1c * W_DIM + x0c) * C_IN;
    const float* p3 = xtb + (size_t)(y1c * W_DIM + x1c) * C_IN;

    for (int cb = 0; cb < 2; ++cb) {
      int kb = kk * 2 + cb;
      int cbase = cb * 32 + khalf * 16;
      float val[16];
      #pragma unroll
      for (int i = 0; i < 16; ++i) val[i] = 0.0f;
      const float* cp[4] = { p0 + cbase, p1 + cbase, p2 + cbase, p3 + cbase };
      float cwv[4] = { cw0, cw1, cw2, cw3 };
      #pragma unroll
      for (int cn = 0; cn < 4; ++cn) {
        const float4* p = (const float4*)cp[cn];
        float wgt = cwv[cn];
        #pragma unroll
        for (int qd = 0; qd < 4; ++qd) {
          float4 q = p[qd];
          val[qd*4+0] += wgt * q.x; val[qd*4+1] += wgt * q.y;
          val[qd*4+2] += wgt * q.z; val[qd*4+3] += wgt * q.w;
        }
      }
      AFrag Bh, Bl;
      #pragma unroll
      for (int v = 0; v < 8; ++v) {
        float a0 = val[2*v], a1 = val[2*v+1];
        Bh.u[v] = pack_h2(a0, a1);
        Bl.u[v] = pack_h2(a0 - (float)(_Float16)a0, a1 - (float)(_Float16)a1);
      }
      #pragma unroll
      for (int m = 0; m < 4; ++m) {
        AFrag Ah, Al;
        const uint4* pH = (const uint4*)(packD + ((((size_t)(0*4 + m) * NKB + kb) * 32 + lane) * 8));
        const uint4* pL = (const uint4*)(packD + ((((size_t)(1*4 + m) * NKB + kb) * 32 + lane) * 8));
        Ah.q[0] = pH[0]; Ah.q[1] = pH[1];
        Al.q[0] = pL[0]; Al.q[1] = pL[1];
        acc[m].v = __builtin_amdgcn_wmma_f32_16x16x32_f16(false, Ah.v, false, Bh.v, (short)0, acc[m].v, false, false);
        acc[m].v = __builtin_amdgcn_wmma_f32_16x16x32_f16(false, Ah.v, false, Bl.v, (short)0, acc[m].v, false, false);
        acc[m].v = __builtin_amdgcn_wmma_f32_16x16x32_f16(false, Al.v, false, Bh.v, (short)0, acc[m].v, false, false);
      }
    }
  }
  #pragma unroll
  for (int m = 0; m < 4; ++m)
    #pragma unroll
    for (int r = 0; r < 8; ++r) {
      int o = m * 16 + r + khalf * 8;
      out[(((size_t)b * COUT + o) * H_DIM + h) * W_DIM + w0 + px] = acc[m].f[r] + db[o];
    }
}

// ---------------------------------------------------------------------------
extern "C" void kernel_launch(void* const* d_in, const int* in_sizes, int n_in,
                              void* d_out, int out_size, void* d_ws, size_t ws_size,
                              hipStream_t stream) {
  const float* x  = (const float*)d_in[0];
  const float* ow = (const float*)d_in[1];
  const float* ob = (const float*)d_in[2];
  const float* dw = (const float*)d_in[3];
  const float* db = (const float*)d_in[4];
  float* out = (float*)d_out;
  int Bn = in_sizes[0] / (C_IN * HW);

  // ws layout: [offsets f32 | x transposed NHWC | packed deform A | packed offset A]
  size_t offBytes = (size_t)Bn * OFFC * HW * sizeof(float);
  size_t xtBytes  = (size_t)Bn * HW * C_IN * sizeof(float);
  float*    off   = (float*)d_ws;
  float*    xtp   = (float*)((char*)d_ws + ((offBytes + 255) & ~(size_t)255));
  unsigned* packD = (unsigned*)((char*)xtp + ((xtBytes + 255) & ~(size_t)255));
  const int ND = 2 * 4 * NKB * 32 * 8;
  const int NO = 2 * 2 * NKB * 32 * 8;
  unsigned* packO = packD + ND;

  prep_pack_weights<<<(ND + NO + 255) / 256, 256, 0, stream>>>(dw, ow, packD, packO);

  dim3 tgrid(Bn * H_DIM, W_DIM / 64);
  transpose_nchw_nhwc<<<tgrid, 256, 0, stream>>>(x, xtp);

  int tiles = Bn * H_DIM * (W_DIM / 16);
  int blocks = (tiles + 3) / 4;
  offset_gemm<<<blocks, 128, 0, stream>>>(xtp, packO, ob, off, Bn);
  deform_gemm<<<blocks, 128, 0, stream>>>(xtp, off, packD, db, out, Bn);
}
